// RGCNLayer_6571299963187
// MI455X (gfx1250) — compile-verified, matrix operand order checked by reading, not run on hardware
//
#include <hip/hip_runtime.h>

typedef __attribute__((ext_vector_type(16))) __bf16 v16bf;
typedef __attribute__((ext_vector_type(8)))  float  v8f;

#define D      128
#define TILE_M 128
#define LDSP   136   // padded LDS row stride in ushorts (272B, 16B-aligned)

__device__ inline unsigned short f2bf(float f) {
  unsigned u = __builtin_bit_cast(unsigned, f);
  u += 0x7FFFu + ((u >> 16) & 1u);        // round-to-nearest-even
  return (unsigned short)(u >> 16);
}
__device__ inline float bf2f(unsigned short b) {
  return __builtin_bit_cast(float, ((unsigned)b) << 16);
}

union V16 { uint4 q[2]; v16bf v; };

// h[m][n] = relu( x[tile_m0+m][:] . Wr[:][n] ), bf16 output, one relation.
__global__ __launch_bounds__(256) void rgcn_transform_kernel(
    const float* __restrict__ x, const float* __restrict__ Wr,
    unsigned short* __restrict__ h, int n_nodes)
{
  __shared__ __align__(16) unsigned short sA[TILE_M * LDSP]; // x tile, bf16
  __shared__ __align__(16) unsigned short sB[D * LDSP];      // Wr transposed, bf16

  const int t = threadIdx.x;
  const int tile_m0 = blockIdx.x * TILE_M;
  const int remain = n_nodes - tile_m0;
  const int rows = remain < TILE_M ? remain : TILE_M;

  // Stage 128x128 x-tile and 128x128 W (transposed) into LDS as bf16.
  // 4096 float4 total -> 16 float4 per thread.
  #pragma unroll
  for (int i = 0; i < 16; ++i) {
    int idx4 = i * 256 + t;
    int r = idx4 >> 5;          // 0..127
    int c = (idx4 & 31) * 4;    // 0..124
    float4 vx = make_float4(0.f, 0.f, 0.f, 0.f);
    if (r < rows) vx = *(const float4*)(x + (size_t)(tile_m0 + r) * D + c);
    sA[r * LDSP + c + 0] = f2bf(vx.x);
    sA[r * LDSP + c + 1] = f2bf(vx.y);
    sA[r * LDSP + c + 2] = f2bf(vx.z);
    sA[r * LDSP + c + 3] = f2bf(vx.w);
    float4 vw = *(const float4*)(Wr + (size_t)r * D + c);   // W[k=r][o=c..c+3]
    sB[(c + 0) * LDSP + r] = f2bf(vw.x);                    // sB[o][k]
    sB[(c + 1) * LDSP + r] = f2bf(vw.y);
    sB[(c + 2) * LDSP + r] = f2bf(vw.z);
    sB[(c + 3) * LDSP + r] = f2bf(vw.w);
  }
  __syncthreads();

  const int wave = t >> 5, lane = t & 31;
  const int wm = (wave & 1) * 64;    // 2x4 wave grid: 64 rows x 32 cols per wave
  const int wn = (wave >> 1) * 32;
  const int lr = lane & 15, g = lane >> 4;

  v8f acc[4][2] = {};
  #pragma unroll
  for (int kb = 0; kb < D; kb += 32) {
    V16 a[4], b[2];
    #pragma unroll
    for (int mi = 0; mi < 4; ++mi) {
      // 16-bit A layout: lane holds row (wm+mi*16+lr); K = {g*8..+7, 16+g*8..+7}
      const unsigned short* p = &sA[(wm + mi * 16 + lr) * LDSP + kb + g * 8];
      a[mi].q[0] = *(const uint4*)p;          // ds_load_b128
      a[mi].q[1] = *(const uint4*)(p + 16);   // +32B => K+16
    }
    #pragma unroll
    for (int ni = 0; ni < 2; ++ni) {
      const unsigned short* p = &sB[(wn + ni * 16 + lr) * LDSP + kb + g * 8];
      b[ni].q[0] = *(const uint4*)p;
      b[ni].q[1] = *(const uint4*)(p + 16);
    }
    #pragma unroll
    for (int mi = 0; mi < 4; ++mi)
      #pragma unroll
      for (int ni = 0; ni < 2; ++ni)
        acc[mi][ni] = __builtin_amdgcn_wmma_f32_16x16x32_bf16(
            false, a[mi].v, false, b[ni].v, (short)0, acc[mi][ni], false, false);
  }

  // C/D layout: VGPR i -> M = i + 8*g ; N = lr. Fuse ReLU, store bf16.
  #pragma unroll
  for (int mi = 0; mi < 4; ++mi)
    #pragma unroll
    for (int ni = 0; ni < 2; ++ni)
      #pragma unroll
      for (int i = 0; i < 8; ++i) {
        int m = tile_m0 + wm + mi * 16 + g * 8 + i;
        int n = wn + ni * 16 + lr;
        if (m < n_nodes) {
          float v = acc[mi][ni][i];
          h[(size_t)m * D + n] = f2bf(v > 0.f ? v : 0.f);
        }
      }
}

__global__ void zero_kernel(float* __restrict__ z, int n) {
  int i = blockIdx.x * blockDim.x + threadIdx.x;
  if (i < n) z[i] = 0.f;
}

// One 32-lane wave per edge; lanes move 4 floats each (128 = 32*4).
__global__ __launch_bounds__(256) void rgcn_scatter_kernel(
    const unsigned short* __restrict__ h,
    const int* __restrict__ edge_dst, const int* __restrict__ edge_src,
    const int* __restrict__ edge_type,
    float* __restrict__ z, int n_edges, int rel)
{
  int gid = blockIdx.x * 256 + threadIdx.x;
  int e = gid >> 5;
  int lane = threadIdx.x & 31;
  if (e >= n_edges) return;
  if (edge_type[e] != rel) return;
  int dst = edge_dst[e];
  int src = edge_src[e];
  uint2 p = *(const uint2*)(h + (size_t)src * D + lane * 4);   // 4 bf16
  float* zp = z + (size_t)dst * D + lane * 4;
  __hip_atomic_fetch_add(zp + 0, bf2f((unsigned short)(p.x & 0xffffu)),
                         __ATOMIC_RELAXED, __HIP_MEMORY_SCOPE_AGENT);
  __hip_atomic_fetch_add(zp + 1, bf2f((unsigned short)(p.x >> 16)),
                         __ATOMIC_RELAXED, __HIP_MEMORY_SCOPE_AGENT);
  __hip_atomic_fetch_add(zp + 2, bf2f((unsigned short)(p.y & 0xffffu)),
                         __ATOMIC_RELAXED, __HIP_MEMORY_SCOPE_AGENT);
  __hip_atomic_fetch_add(zp + 3, bf2f((unsigned short)(p.y >> 16)),
                         __ATOMIC_RELAXED, __HIP_MEMORY_SCOPE_AGENT);
}

extern "C" void kernel_launch(void* const* d_in, const int* in_sizes, int n_in,
                              void* d_out, int out_size, void* d_ws, size_t ws_size,
                              hipStream_t stream) {
  const float* x  = (const float*)d_in[0];   // [N, 128] f32
  const float* W  = (const float*)d_in[1];   // [R, 128, 128] f32
  const int*   ei = (const int*)d_in[2];     // [2, E] i32: [0]=dst, [1]=src
  const int*   et = (const int*)d_in[3];     // [E] i32
  float* z = (float*)d_out;                  // [N, 128] f32
  unsigned short* h = (unsigned short*)d_ws; // [N, 128] bf16 (25.6 MB, L2-resident)

  const int E    = in_sizes[3];
  const int Nn   = in_sizes[0] / D;
  const int NREL = in_sizes[1] / (D * D);

  zero_kernel<<<(out_size + 255) / 256, 256, 0, stream>>>(z, out_size);

  const int mtiles  = (Nn + TILE_M - 1) / TILE_M;
  const int sblocks = (E * 32 + 255) / 256;
  for (int r = 0; r < NREL; ++r) {
    rgcn_transform_kernel<<<mtiles, 256, 0, stream>>>(
        x, W + (size_t)r * D * D, h, Nn);
    rgcn_scatter_kernel<<<sblocks, 256, 0, stream>>>(
        h, ei, ei + E, et, z, E, r);
  }
}
